// TemporalConvolution_26328149524715
// MI455X (gfx1250) — matrix operand order repaired and checked by reading, not compiled, not run on hardware
//
#include <hip/hip_runtime.h>
#include <cstdint>

typedef __attribute__((ext_vector_type(16))) __bf16 v16bf;
typedef __attribute__((ext_vector_type(8)))  __bf16 v8bf;
typedef __attribute__((ext_vector_type(8)))  float  v8f;

#define BATCH 32
#define CIN   192
#define LIN   8192
#define COUT  384
#define KW    9
#define LOUT  8183
#define KK    (CIN*KW)        // 1728
#define TM    64              // l tile
#define TN    64              // cout tile
#define WIN   (TM + KW - 1)   // 72
#define XS    200             // xwin row stride (bf16): 400B rows -> 16B aligned, bank-spread
#define OS    68              // epilogue staging row stride (f32)

// ---- one-time weight prep: f32 [cout][cin][t] -> bf16 hi/lo at k' = t*192+cin ----
__global__ __launch_bounds__(256)
void prep_weights(const float* __restrict__ w,
                  __bf16* __restrict__ whi, __bf16* __restrict__ wlo)
{
    int flat = blockIdx.x * 256 + threadIdx.x;   // cout*1728 + cin*9 + t
    if (flat >= COUT * KK) return;
    int cout = flat / KK;
    int rem  = flat - cout * KK;
    int cin  = rem / KW;
    int t    = rem - cin * KW;
    float v  = w[flat];
    __bf16 h = (__bf16)v;
    __bf16 l = (__bf16)(v - (float)h);
    size_t o = (size_t)cout * KK + t * CIN + cin;
    whi[o] = h;
    wlo[o] = l;
}

// ---- main conv kernel: implicit GEMM, bf16 hi/lo 3-product WMMA ----
__global__ __launch_bounds__(256)
void conv1d_wmma(const float*  __restrict__ x,
                 const __bf16* __restrict__ whi,
                 const __bf16* __restrict__ wlo,
                 const float*  __restrict__ bias,
                 float*        __restrict__ out)
{
    // x window hi/lo, transposed [pos][cin]: 2 * 72*200 bf16 = 57,600 B
    __shared__ __align__(32) unsigned char smem_raw[2 * WIN * XS * 2];
    __bf16* xh   = (__bf16*)smem_raw;
    __bf16* xl   = xh + WIN * XS;
    float*  obuf = (float*)smem_raw;     // reused after K loop (17,408 B fits)

    const int tid = threadIdx.x;
    const int l0  = blockIdx.x * TM;
    const int n0  = blockIdx.y * TN;
    const int b   = blockIdx.z;

    // ---- stage + decompose x window once (coalesced global reads) ----
    for (int idx = tid; idx < CIN * WIN; idx += 256) {
        int cin = idx / WIN;
        int p   = idx - cin * WIN;
        int pos = l0 + p;
        float v = 0.0f;
        if (pos < LIN) v = x[((size_t)b * CIN + cin) * LIN + pos];
        __bf16 h = (__bf16)v;
        xh[p * XS + cin] = h;
        xl[p * XS + cin] = (__bf16)(v - (float)h);
    }
    __syncthreads();

    const int lane = tid & 31;
    const int wave = tid >> 5;
    const int half = lane >> 4;          // wave32 half (ISA fragment layouts)
    const int l16  = lane & 15;
    const int mw   = wave & 3;           // 4 M-subtiles of 16
    const int nw   = wave >> 2;          // 2 N-subtiles of 32
    const int mrow = mw * 16 + l16;      // A row within the 64-l tile
    const int ncol = n0 + nw * 32 + l16; // B column for s=0 (s=1 -> +16)

    v8f acc0 = {};
    v8f acc1 = {};

    // Incrementally-advanced base pointers: inner offsets are compile-time
    // immediates (c*64B), outer loop is two pointer adds per tap.
    const __bf16* arh = xh + mrow * XS + 16 * half;         // A hi row (tap 0)
    const __bf16* arl = xl + mrow * XS + 16 * half;         // A lo row (tap 0)
    const __bf16* wht = whi + (size_t)ncol * KK + 16 * half; // B hi col s=0 (tap 0)
    const __bf16* wlt = wlo + (size_t)ncol * KK + 16 * half; // B lo col s=0 (tap 0)

    for (int t = 0; t < KW; ++t) {
        if (t + 1 < KW)
            __builtin_prefetch(wht + CIN, 0, 1);   // next tap's weights -> global_prefetch_b8

        #pragma unroll
        for (int c = 0; c < 6; ++c) {
            // A fragments: two contiguous 16B LDS loads each (ds_load_b128, imm offsets)
            const v8bf* ah = (const v8bf*)(arh + c * 32);
            const v8bf* al = (const v8bf*)(arl + c * 32);
            v16bf ahi = __builtin_shufflevector(ah[0], ah[1],
                           0,1,2,3,4,5,6,7,8,9,10,11,12,13,14,15);
            v16bf alo = __builtin_shufflevector(al[0], al[1],
                           0,1,2,3,4,5,6,7,8,9,10,11,12,13,14,15);

            // B fragments: contiguous 32B global loads (L1/L2-resident weights)
            v16bf b0h = *(const v16bf*)(wht + c * 32);
            v16bf b0l = *(const v16bf*)(wlt + c * 32);
            v16bf b1h = *(const v16bf*)(wht + (size_t)16 * KK + c * 32); // col ncol+16
            v16bf b1l = *(const v16bf*)(wlt + (size_t)16 * KK + c * 32);

            // hi/lo split products, f32 accumulate (near-f32 accuracy)
            acc0 = __builtin_amdgcn_wmma_f32_16x16x32_bf16(false, ahi, false, b0h, (short)0, acc0, false, false);
            acc0 = __builtin_amdgcn_wmma_f32_16x16x32_bf16(false, ahi, false, b0l, (short)0, acc0, false, false);
            acc0 = __builtin_amdgcn_wmma_f32_16x16x32_bf16(false, alo, false, b0h, (short)0, acc0, false, false);
            acc1 = __builtin_amdgcn_wmma_f32_16x16x32_bf16(false, ahi, false, b1h, (short)0, acc1, false, false);
            acc1 = __builtin_amdgcn_wmma_f32_16x16x32_bf16(false, ahi, false, b1l, (short)0, acc1, false, false);
            acc1 = __builtin_amdgcn_wmma_f32_16x16x32_bf16(false, alo, false, b1h, (short)0, acc1, false, false);
        }

        arh += XS;    // +400 B: next tap's x row
        arl += XS;
        wht += CIN;   // +384 B: next tap's weight segment
        wlt += CIN;
    }

    __syncthreads();   // x window done -> reuse LDS as [cout][l] f32 staging
    #pragma unroll
    for (int s = 0; s < 2; ++s) {
        v8f c = s ? acc1 : acc0;
        const int n = nw * 32 + s * 16 + l16;
        #pragma unroll
        for (int r = 0; r < 8; ++r) {
            int m = mw * 16 + r + 8 * half;   // C layout: VGPR r -> M=r (+8 upper half)
            obuf[n * OS + m] = c[r];
        }
    }
    __syncthreads();

    // ---- coalesced epilogue: bias add + bounds-checked store ----
    #pragma unroll
    for (int p = 0; p < 16; ++p) {
        int n = (tid >> 6) + 4 * p;   // 4 cout rows per pass
        int l = tid & 63;             // contiguous l across the wave
        float v = obuf[n * OS + l] + bias[n0 + n];
        if (l0 + l < LOUT)
            out[((size_t)b * COUT + (n0 + n)) * LOUT + l0 + l] = v;
    }
}

extern "C" void kernel_launch(void* const* d_in, const int* in_sizes, int n_in,
                              void* d_out, int out_size, void* d_ws, size_t ws_size,
                              hipStream_t stream) {
    const float* x    = (const float*)d_in[0];
    const float* w    = (const float*)d_in[1];
    const float* bias = (const float*)d_in[2];
    float* out        = (float*)d_out;
    (void)in_sizes; (void)n_in; (void)out_size; (void)ws_size;

    // workspace: decomposed+reordered weights, 2 * 384*1728 bf16 = 2,654,208 B
    __bf16* whi = (__bf16*)d_ws;
    __bf16* wlo = whi + (size_t)COUT * KK;

    prep_weights<<<(COUT * KK) / 256, 256, 0, stream>>>(w, whi, wlo);

    dim3 grid(128 /* ceil(8183/64) */, COUT / TN /* 6 */, BATCH /* 32 */);
    conv1d_wmma<<<grid, 256, 0, stream>>>(x, whi, wlo, bias, out);
}
